// mGRU_71416716198197
// MI455X (gfx1250) — matrix operand order repaired and checked by reading, not compiled, not run on hardware
//
#include <hip/hip_runtime.h>

typedef unsigned short u16;
typedef __attribute__((ext_vector_type(16))) __bf16 v16bf;
typedef __attribute__((ext_vector_type(8)))  float  v8f;

// ---------- bf16 helpers ----------
__device__ __forceinline__ float bf2f(u16 u) {
  union { unsigned i; float f; } v; v.i = ((unsigned)u) << 16; return v.f;
}
__device__ __forceinline__ u16 f2bf(float f) {
  union { float f; unsigned i; } v; v.f = f;
  unsigned x = v.i;
  x += 0x7FFFu + ((x >> 16) & 1u);       // round-to-nearest-even
  return (u16)(x >> 16);
}
__device__ __forceinline__ float sigf(float x) { return 1.f / (1.f + __expf(-x)); }

// A-operand (16x32 bf16) packed-layout index for element (m,k) of an MxK matrix.
// Layout per CDNA5 ISA 7.12.2: lane = khalf*16 + (m&15);
// VGPR j<4 : k = khalf*8 + 2j+p ; VGPR j>=4 : k = 16 + khalf*8 + 2(j-4)+p
__device__ __forceinline__ size_t apk_idx(int m, int k, int KT32) {
  int kt   = k >> 5;
  int kin  = k & 31;
  int hi   = kin >> 4;
  int rem  = kin & 15;
  int khalf = rem >> 3;
  int low3  = rem & 7;
  int j    = ((((hi << 2) | (low3 >> 1)) << 1) | (low3 & 1));
  int lane = (khalf << 4) | (m & 15);
  return (((size_t)(m >> 4) * KT32 + kt) * 32 + lane) * 16 + j;
}

__device__ __forceinline__ v8f wmma_bf(const u16* ap, const u16* bp, v8f c) {
  v16bf a = *(const v16bf*)ap;
  v16bf b = *(const v16bf*)bp;
  return __builtin_amdgcn_wmma_f32_16x16x32_bf16(false, a, false, b, (short)0, c,
                                                 false, false);
}

// ---------- CDNA5 async LDS->global (ASYNCcnt path) ----------
__device__ __forceinline__ void async_store_from_lds_b128(void* gaddr, unsigned lds_off) {
  asm volatile("global_store_async_from_lds_b128 %0, %1, off"
               :: "v"(gaddr), "v"(lds_off) : "memory");
}
__device__ __forceinline__ void wait_asynccnt0() {
  asm volatile("s_wait_asynccnt 0x0" ::: "memory");
}

// ---------- packing kernels ----------
__global__ void pack_a_kernel(const float* __restrict__ src, u16* __restrict__ dst,
                              long long total, int K) {
  long long idx = (long long)blockIdx.x * blockDim.x + threadIdx.x;
  if (idx >= total) return;
  int m = (int)(idx / K);
  int k = (int)(idx % K);
  dst[apk_idx(m, k, K >> 5)] = f2bf(src[idx]);
}

// Pack B = src (KxN) or src^T (src is NxK) into WMMA B-operand (32x16) layout.
__global__ void pack_b_kernel(const float* __restrict__ src, u16* __restrict__ dst,
                              int K, int N, int ld, int transposed, long long total) {
  long long idx = (long long)blockIdx.x * blockDim.x + threadIdx.x;
  if (idx >= total) return;
  int k = (int)(idx / N);
  int n = (int)(idx % N);
  float v = transposed ? src[(size_t)n * ld + k] : src[(size_t)k * ld + n];
  int kt = k >> 5, kin = k & 31;
  int khalf = (kin >> 3) & 1, j = ((kin >> 4) << 3) | (kin & 7);
  int nt = n >> 4, NT = N >> 4;
  int lane = (khalf << 4) | (n & 15);
  dst[(((size_t)kt * NT + nt) * 32 + lane) * 16 + j] = f2bf(v);
}

// ---------- generic bf16 WMMA GEMM: C = A @ B (+bias) ----------
__global__ void gemm_bf16_kernel(const u16* __restrict__ Ap, const u16* __restrict__ Bp,
                                 const float* __restrict__ bias,
                                 float* __restrict__ Cf, u16* __restrict__ Crm,
                                 u16* __restrict__ Capk, int M, int N, int K) {
  const int lane = threadIdx.x & 31;
  const int w = blockIdx.x * (blockDim.x >> 5) + (threadIdx.x >> 5);
  const int NT = N >> 4, MT = M >> 4, KT = K >> 5;
  if (w >= MT * NT) return;
  const int mt = w / NT, nt = w % NT;
  const int col = lane & 15, hlf = lane >> 4;

  v8f c;
  {
    float bv = bias ? bias[nt * 16 + col] : 0.f;
    for (int r = 0; r < 8; ++r) c[r] = bv;
  }
  const u16* arow = Ap + (size_t)mt * KT * 512;   // each 16x32 tile = 512 halfs
  const u16* bcol = Bp + (size_t)nt * 512;
  for (int kt = 0; kt < KT; ++kt) {
    __builtin_prefetch(arow + (size_t)(kt + 2) * 512 + lane * 16, 0, 1);
    __builtin_prefetch(bcol + (size_t)(kt + 2) * NT * 512 + lane * 16, 0, 1);
    c = wmma_bf(arow + (size_t)kt * 512 + lane * 16,
                bcol + (size_t)kt * NT * 512 + lane * 16, c);
  }
  for (int r = 0; r < 8; ++r) {
    int m = mt * 16 + hlf * 8 + r;
    int n = nt * 16 + col;
    float v = c[r];
    if (Cf)   Cf[(size_t)m * N + n] = v;
    if (Crm)  Crm[(size_t)m * N + n] = f2bf(v);
    if (Capk) Capk[apk_idx(m, n, N >> 5)] = f2bf(v);
  }
}

// ---------- persistent GRU scan ----------
// One block per GRU stream. h (128x512 bf16, A-operand layout) is double-buffered
// in LDS; per step each wave owns 8 (r,z,n) gate triples and updates h straight
// from the WMMA C-layout registers. Step output drains to global via
// global_store_async_from_lds_b128 (ASYNCcnt).
__global__ void gru_scan_kernel(
    const u16* gi0, const u16* whh0, const float* bhh0, u16* oapk0, u16* orm0,
    const u16* gi1, const u16* whh1, const float* bhh1, u16* oapk1, u16* orm1,
    int T) {
  extern __shared__ u16 smem[];
  u16* cur = smem;            // 65536 halfs = 128KB
  u16* nxt = smem + 65536;    // 128KB
  const u16* gi;  const u16* whh; const float* bhh; u16* oapk; u16* orm;
  if (blockIdx.x == 0) { gi = gi0; whh = whh0; bhh = bhh0; oapk = oapk0; orm = orm0; }
  else                 { gi = gi1; whh = whh1; bhh = bhh1; oapk = oapk1; orm = orm1; }

  const int tid = threadIdx.x;
  const int lane = tid & 31, wave = tid >> 5;
  const int col = lane & 15, hlf = lane >> 4;

  for (int i = tid; i < 128 * 512; i += 1024) cur[i] = 0x3F80u;   // h0 = 1.0
  __syncthreads();

  for (int t = 0; t < T; ++t) {
    // phase 1: gh = h @ Whh^T + bhh fused with GRU gate update -> nxt (LDS only)
    for (int q = wave * 8; q < wave * 8 + 8; ++q) {
      int mt = q >> 5, dt = q & 31;
      int n0 = dt * 16 + col;
      v8f cr, cz, cn;
      {
        float br = bhh[n0], bz = bhh[512 + n0], bn = bhh[1024 + n0];
        for (int r = 0; r < 8; ++r) { cr[r] = br; cz[r] = bz; cn[r] = bn; }
      }
      for (int kt = 0; kt < 16; ++kt) {
        const u16* ap = &cur[((mt * 16 + kt) * 32 + lane) * 16];
        const u16* bp = &whh[(size_t)((kt * 96 + dt) * 32 + lane) * 16];
        __builtin_prefetch(bp + 96 * 512, 0, 1);   // next kt, gate r
        cr = wmma_bf(ap, bp, cr);
        cz = wmma_bf(ap, bp + 32 * 512, cz);       // gate z tiles: nt += 32
        cn = wmma_bf(ap, bp + 64 * 512, cn);       // gate n tiles: nt += 64
      }
      for (int r = 0; r < 8; ++r) {
        int b = mt * 16 + hlf * 8 + r;
        int d = n0;
        size_t gb = ((size_t)(t * 128 + b)) * 1536 + d;
        float rg = sigf(bf2f(gi[gb]) + cr[r]);
        float zg = sigf(bf2f(gi[gb + 512]) + cz[r]);
        float ng = tanhf(bf2f(gi[gb + 1024]) + rg * cn[r]);
        float ho = bf2f(cur[apk_idx(b, d, 16)]);
        nxt[apk_idx(b, d, 16)] = f2bf((1.f - zg) * ng + zg * ho);
      }
    }
    __syncthreads();   // nxt complete (waits DScnt)

    // phase 2a: bulk drain nxt -> o_apk via CDNA5 async LDS->global b128
    for (int i = tid; i < 8192; i += 1024) {
      unsigned loff = (unsigned)(unsigned long long)(nxt + i * 8);
      void* g = (void*)(oapk + (size_t)t * 65536 + (size_t)i * 8);
      async_store_from_lds_b128(g, loff);
    }
    // phase 2b: row-major bf16 copy for the attention sweep (premise only)
    if (orm) {
      for (int i = tid; i < 8192; i += 1024) {
        int m = i >> 6, d0 = (i & 63) * 8;
        union { u16 h[8]; uint4 v; } u;
        for (int j = 0; j < 8; ++j) u.h[j] = nxt[apk_idx(m, d0 + j, 16)];
        *reinterpret_cast<uint4*>(orm + ((size_t)(t * 128 + m)) * 512 + d0) = u.v;
      }
    }
    wait_asynccnt0();
    __syncthreads();
    u16* tmp = cur; cur = nxt; nxt = tmp;
  }
}

// ---------- persistent attention + match-GRU scan ----------
__global__ void match_scan_kernel(
    const u16* __restrict__ oh_apk, const u16* __restrict__ whs,
    const u16* __restrict__ op_rm, const float* __restrict__ we,
    const u16* __restrict__ wtwm, const u16* __restrict__ mwih,
    const u16* __restrict__ mwhh, const float* __restrict__ m_bih,
    const float* __restrict__ m_bhh, float* __restrict__ logitsG,
    float* __restrict__ giM, float* __restrict__ ghM,
    const float* __restrict__ outW, const float* __restrict__ outB,
    float* __restrict__ outp, int T) {
  extern __shared__ u16 smem[];
  u16* hmA = smem;           // h_m, A-packed bf16, 65536 halfs
  u16* UB  = smem + 65536;   // u = Wht+Wm (rm), later a_t (A-packed)

  const int tid = threadIdx.x, lane = tid & 31, wave = tid >> 5;
  const int col = lane & 15, hlf = lane >> 4;

  for (int i = tid; i < 65536; i += 1024) hmA[i] = 0x3F80u;   // h_m0 = 1.0
  __syncthreads();

  for (int t = 0; t < T; ++t) {
    // A: u = [o_h[t] | h_m] @ [W_t ; W_m]   (M=128,N=512,K=1024)
    for (int q = wave * 8; q < wave * 8 + 8; ++q) {
      int mt = q >> 5, nt = q & 31;
      v8f c; for (int r = 0; r < 8; ++r) c[r] = 0.f;
      for (int kt = 0; kt < 32; ++kt) {
        const u16* ap = (kt < 16)
            ? &oh_apk[(size_t)(((t * 8 + mt) * 16 + kt) * 32 + lane) * 16]
            : &hmA[((mt * 16 + (kt - 16)) * 32 + lane) * 16];
        const u16* bp = &wtwm[(size_t)((kt * 32 + nt) * 32 + lane) * 16];
        __builtin_prefetch(bp + 32 * 512, 0, 1);
        c = wmma_bf(ap, bp, c);
      }
      for (int r = 0; r < 8; ++r) {
        int b = mt * 16 + hlf * 8 + r, d = nt * 16 + col;
        UB[b * 512 + d] = f2bf(c[r]);
      }
    }
    __syncthreads();

    // B: logits[b][tp] = sum_d tanh(Whs[tp,b,d] + u[b,d]) * we[d]
    //    wave per (tp,b); lane owns 16 consecutive d -> 2x b128 per operand
    for (int pair = wave; pair < 256 * 128; pair += 32) {
      int tp = pair >> 7, b = pair & 127;
      const u16* wr = &whs[((size_t)tp * 128 + b) * 512];
      const u16* ur = &UB[b * 512];
      int d0 = lane * 16;
      union { uint4 v; u16 h[8]; } wa, wb, ua, ub;
      wa.v = *(const uint4*)(wr + d0);
      wb.v = *(const uint4*)(wr + d0 + 8);
      ua.v = *(const uint4*)(ur + d0);
      ub.v = *(const uint4*)(ur + d0 + 8);
      float acc = 0.f;
      for (int j = 0; j < 8; ++j)
        acc += tanhf(bf2f(wa.h[j]) + bf2f(ua.h[j])) * we[d0 + j];
      for (int j = 0; j < 8; ++j)
        acc += tanhf(bf2f(wb.h[j]) + bf2f(ub.h[j])) * we[d0 + 8 + j];
      for (int o = 16; o > 0; o >>= 1) acc += __shfl_xor(acc, o, 32);
      if (lane == 0) logitsG[b * 256 + tp] = acc;
    }
    __threadfence();
    __syncthreads();

    // C: softmax over tp, per b (wave per row)
    for (int b = wave; b < 128; b += 32) {
      float* row = &logitsG[b * 256];
      float mx = -3.0e38f;
      for (int i = lane; i < 256; i += 32) mx = fmaxf(mx, row[i]);
      for (int o = 16; o > 0; o >>= 1) mx = fmaxf(mx, __shfl_xor(mx, o, 32));
      float s = 0.f;
      for (int i = lane; i < 256; i += 32) { float e = __expf(row[i] - mx); row[i] = e; s += e; }
      for (int o = 16; o > 0; o >>= 1) s += __shfl_xor(s, o, 32);
      float inv = 1.f / s;
      for (int i = lane; i < 256; i += 32) row[i] *= inv;
    }
    __threadfence();
    __syncthreads();

    // D: a_t[b][d] = sum_tp alpha[b][tp] * o_p[tp][b][d]  -> UB as A-packed
    //    thread owns 8 consecutive d -> uint4 loads of o_p
    for (int cI = tid; cI < 8192; cI += 1024) {
      int b = cI >> 6, d0 = (cI & 63) * 8;
      float acc[8];
      for (int j = 0; j < 8; ++j) acc[j] = 0.f;
      const float* al = &logitsG[b * 256];
      for (int tp = 0; tp < 256; ++tp) {
        float a = al[tp];
        union { uint4 v; u16 h[8]; } o;
        o.v = *(const uint4*)(op_rm + ((size_t)tp * 128 + b) * 512 + d0);
        for (int j = 0; j < 8; ++j) acc[j] += a * bf2f(o.h[j]);
      }
      for (int j = 0; j < 8; ++j) UB[apk_idx(b, d0 + j, 16)] = f2bf(acc[j]);
    }
    __syncthreads();

    // E: gi_m = [a_t | h_t] @ m_Wih^T + m_bih   (K=1024, N=1536)
    for (int q = wave; q < 8 * 96; q += 32) {
      int mt = q / 96, nt = q % 96;
      int n0 = nt * 16 + col;
      v8f c; { float bv = m_bih[n0]; for (int r = 0; r < 8; ++r) c[r] = bv; }
      for (int kt = 0; kt < 32; ++kt) {
        const u16* ap = (kt < 16)
            ? &UB[((mt * 16 + kt) * 32 + lane) * 16]
            : &oh_apk[(size_t)(((t * 8 + mt) * 16 + (kt - 16)) * 32 + lane) * 16];
        const u16* bp = &mwih[(size_t)((kt * 96 + nt) * 32 + lane) * 16];
        __builtin_prefetch(bp + 96 * 512, 0, 1);
        c = wmma_bf(ap, bp, c);
      }
      for (int r = 0; r < 8; ++r)
        giM[(size_t)(mt * 16 + hlf * 8 + r) * 1536 + n0] = c[r];
    }
    // F: gh_m = h_m @ m_Whh^T + m_bhh   (K=512, N=1536)
    for (int q = wave; q < 8 * 96; q += 32) {
      int mt = q / 96, nt = q % 96;
      int n0 = nt * 16 + col;
      v8f c; { float bv = m_bhh[n0]; for (int r = 0; r < 8; ++r) c[r] = bv; }
      for (int kt = 0; kt < 16; ++kt) {
        const u16* ap = &hmA[((mt * 16 + kt) * 32 + lane) * 16];
        const u16* bp = &mwhh[(size_t)((kt * 96 + nt) * 32 + lane) * 16];
        __builtin_prefetch(bp + 96 * 512, 0, 1);
        c = wmma_bf(ap, bp, c);
      }
      for (int r = 0; r < 8; ++r)
        ghM[(size_t)(mt * 16 + hlf * 8 + r) * 1536 + n0] = c[r];
    }
    __threadfence();
    __syncthreads();

    // G: GRU update of h_m, in place in hmA
    for (int idx = tid; idx < 65536; idx += 1024) {
      int b = idx >> 9, d = idx & 511;
      size_t o = (size_t)b * 1536 + d;
      float rg = sigf(giM[o] + ghM[o]);
      float zg = sigf(giM[o + 512] + ghM[o + 512]);
      float ng = tanhf(giM[o + 1024] + rg * ghM[o + 1024]);
      size_t hp = apk_idx(b, d, 16);
      float ho = bf2f(hmA[hp]);
      hmA[hp] = f2bf((1.f - zg) * ng + zg * ho);
    }
    __syncthreads();
  }

  // out = relu(h_star @ out_W^T + out_b)   (128x3)
  if (tid < 128 * 3) {
    int b = tid / 3, c = tid % 3;
    float acc = outB[c];
    for (int d = 0; d < 512; ++d)
      acc += bf2f(hmA[apk_idx(b, d, 16)]) * outW[c * 512 + d];
    outp[b * 3 + c] = fmaxf(acc, 0.f);
  }
}

// ---------- host ----------
extern "C" void kernel_launch(void* const* d_in, const int* in_sizes, int n_in,
                              void* d_out, int out_size, void* d_ws, size_t ws_size,
                              hipStream_t stream) {
  const float* enc_p  = (const float*)d_in[0];
  const float* enc_h  = (const float*)d_in[1];
  const float* p_W    = (const float*)d_in[2];
  const float* p_b    = (const float*)d_in[3];
  const float* h_W    = (const float*)d_in[4];
  const float* h_b    = (const float*)d_in[5];
  const float* pg_Wih = (const float*)d_in[6];
  const float* pg_Whh = (const float*)d_in[7];
  const float* pg_bih = (const float*)d_in[8];
  const float* pg_bhh = (const float*)d_in[9];
  const float* hg_Wih = (const float*)d_in[10];
  const float* hg_Whh = (const float*)d_in[11];
  const float* hg_bih = (const float*)d_in[12];
  const float* hg_bhh = (const float*)d_in[13];
  const float* W_s    = (const float*)d_in[14];
  const float* W_t    = (const float*)d_in[15];
  const float* w_e    = (const float*)d_in[16];
  const float* W_m    = (const float*)d_in[17];
  const float* m_Wih  = (const float*)d_in[18];
  const float* m_Whh  = (const float*)d_in[19];
  const float* m_bih  = (const float*)d_in[20];
  const float* m_bhh  = (const float*)d_in[21];
  const float* out_W  = (const float*)d_in[22];
  const float* out_b  = (const float*)d_in[23];
  float* outp = (float*)d_out;
  (void)in_sizes; (void)n_in; (void)out_size; (void)ws_size;

  const int Tt = 256, IN = 768, E = 512, H = 512, G3 = 1536;
  const long long M = 256LL * 128LL;   // 32768 rows

  char* ws = (char*)d_ws;
  size_t off = 0;
  auto alloc = [&](size_t bytes) -> char* {
    char* p = ws + off;
    off += (bytes + 255) & ~(size_t)255;
    return p;
  };
  u16* encp_apk = (u16*)alloc((size_t)M * IN * 2);
  u16* ench_apk = (u16*)alloc((size_t)M * IN * 2);
  u16* xp_apk   = (u16*)alloc((size_t)M * E * 2);
  u16* xh_apk   = (u16*)alloc((size_t)M * E * 2);
  u16* gi_p     = (u16*)alloc((size_t)M * G3 * 2);
  u16* gi_h     = (u16*)alloc((size_t)M * G3 * 2);
  u16* op_apk   = (u16*)alloc((size_t)M * H * 2);
  u16* op_rm    = (u16*)alloc((size_t)M * H * 2);
  u16* oh_apk   = (u16*)alloc((size_t)M * H * 2);
  u16* whs      = (u16*)alloc((size_t)M * H * 2);
  u16* pW_pk    = (u16*)alloc((size_t)IN * E * 2);
  u16* hW_pk    = (u16*)alloc((size_t)IN * E * 2);
  u16* pgWih_pk = (u16*)alloc((size_t)E * G3 * 2);
  u16* pgWhh_pk = (u16*)alloc((size_t)H * G3 * 2);
  u16* hgWih_pk = (u16*)alloc((size_t)E * G3 * 2);
  u16* hgWhh_pk = (u16*)alloc((size_t)H * G3 * 2);
  u16* Ws_pk    = (u16*)alloc((size_t)H * H * 2);
  u16* WtWm_pk  = (u16*)alloc((size_t)1024 * H * 2);
  u16* mWih_pk  = (u16*)alloc((size_t)1024 * G3 * 2);
  u16* mWhh_pk  = (u16*)alloc((size_t)H * G3 * 2);
  float* logitsG = (float*)alloc((size_t)128 * 256 * 4);
  float* giM     = (float*)alloc((size_t)128 * 1536 * 4);
  float* ghM     = (float*)alloc((size_t)128 * 1536 * 4);

  hipFuncSetAttribute(reinterpret_cast<const void*>(gru_scan_kernel),
                      hipFuncAttributeMaxDynamicSharedMemorySize, 262144);
  hipFuncSetAttribute(reinterpret_cast<const void*>(match_scan_kernel),
                      hipFuncAttributeMaxDynamicSharedMemorySize, 262144);

  auto packB = [&](const float* src, u16* dst, int K, int N, int ld, int tr) {
    long long tot = (long long)K * N;
    pack_b_kernel<<<(int)((tot + 255) / 256), 256, 0, stream>>>(src, dst, K, N, ld, tr, tot);
  };
  packB(p_W,    pW_pk,    IN,   E,  IN,   1);   // B = p_W^T
  packB(h_W,    hW_pk,    IN,   E,  IN,   1);
  packB(pg_Wih, pgWih_pk, E,    G3, E,    1);
  packB(pg_Whh, pgWhh_pk, H,    G3, H,    1);
  packB(hg_Wih, hgWih_pk, E,    G3, E,    1);
  packB(hg_Whh, hgWhh_pk, H,    G3, H,    1);
  packB(W_s,    Ws_pk,    H,    H,  H,    0);   // direct K x N
  packB(W_t,    WtWm_pk,  H,    H,  H,    0);                       // kt 0..15
  packB(W_m,    WtWm_pk + (size_t)16 * 32 * 512, H, H, H, 0);       // kt 16..31
  packB(m_Wih,  mWih_pk,  1024, G3, 1024, 1);
  packB(m_Whh,  mWhh_pk,  H,    G3, H,    1);

  {
    long long tot = M * IN;
    pack_a_kernel<<<(int)((tot + 255) / 256), 256, 0, stream>>>(enc_p, encp_apk, tot, IN);
    pack_a_kernel<<<(int)((tot + 255) / 256), 256, 0, stream>>>(enc_h, ench_apk, tot, IN);
  }

  auto gemm = [&](const u16* Ap, const u16* Bp, const float* bias,
                  float* Cf, u16* Crm, u16* Capk, int Mm, int Nn, int Kk) {
    long long tiles = (long long)(Mm / 16) * (Nn / 16);
    gemm_bf16_kernel<<<(int)((tiles + 7) / 8), 256, 0, stream>>>(Ap, Bp, bias, Cf, Crm,
                                                                 Capk, Mm, Nn, Kk);
  };
  // x = enc @ W^T + b   (emit A-packed for the next GEMM)
  gemm(encp_apk, pW_pk, p_b, nullptr, nullptr, xp_apk, (int)M, E, IN);
  gemm(ench_apk, hW_pk, h_b, nullptr, nullptr, xh_apk, (int)M, E, IN);
  // gi = x @ Wih^T + bih  (bf16 row-major, streamed by the scans)
  gemm(xp_apk, pgWih_pk, pg_bih, nullptr, gi_p, nullptr, (int)M, G3, E);
  gemm(xh_apk, hgWih_pk, hg_bih, nullptr, gi_h, nullptr, (int)M, G3, E);

  // both GRU scans concurrently (block 0 = premise, block 1 = hypothesis)
  gru_scan_kernel<<<2, 1024, 262144, stream>>>(
      gi_p, pgWhh_pk, pg_bhh, op_apk, op_rm,
      gi_h, hgWhh_pk, hg_bhh, oh_apk, nullptr, Tt);

  // Whs = o_p @ W_s  (bf16 row-major; L2-resident for the attention sweep)
  gemm(op_apk, Ws_pk, nullptr, nullptr, whs, nullptr, (int)M, H, H);

  match_scan_kernel<<<1, 1024, 262144, stream>>>(
      oh_apk, whs, op_rm, w_e, WtWm_pk, mWih_pk, mWhh_pk,
      m_bih, m_bhh, logitsG, giM, ghM, out_W, out_b, outp, Tt);
}
